// NeuralNetworkSimplified_7834020348428
// MI455X (gfx1250) — 8-bit matrix operand order repaired and checked by reading, compiled, not run on hardware
//
#include <hip/hip_runtime.h>
#include <stdint.h>

typedef __attribute__((ext_vector_type(8))) int v8i;

#define D_IN   4096
#define HDIM   500
#define HPAD   512
#define CDIM   12
#define B_ROWS 16384
#define M_TILE 32

#define OFF_W1FRAG 0
#define SZ_W1FRAG  (32 * 64 * 1024)
#define OFF_W2FRAG (OFF_W1FRAG + SZ_W1FRAG)
#define SZ_W2FRAG  (8 * 1024)
#define OFF_S      (OFF_W2FRAG + SZ_W2FRAG)
#define OFF_T      (OFF_S + HPAD * 4)
#define OFF_B2     (OFF_T + HPAD * 4)

__global__ void pack_w1_kernel(const float* __restrict__ W1,
                               uint8_t* __restrict__ w1frag) {
  int t    = blockIdx.x * blockDim.x + threadIdx.x;
  int lane = t & 31;
  int blk  = t >> 5;
  int nt   = blk >> 6;
  int kt   = blk & 63;
  int nn   = lane & 15, hi = lane >> 4;
  int n    = nt * 16 + nn;
  uint32_t* dst = (uint32_t*)(w1frag + (size_t)blk * 1024) + lane * 8;
  for (int v = 0; v < 8; ++v) {
    int half = v >> 2, vk = v & 3;
    uint32_t packed = 0;
    for (int b = 0; b < 4; ++b) {
      int k = kt * 64 + half * 32 + (vk >> 1) * 16 + hi * 8 + (vk & 1) * 4 + b;
      uint32_t sb = 0;
      if (n < HDIM) sb = (W1[(size_t)n * D_IN + k] >= 0.f) ? 0x01u : 0xFFu;
      packed |= sb << (8 * b);
    }
    dst[v] = packed;
  }
}

__global__ void pack_w2_kernel(const float* __restrict__ W2,
                               uint8_t* __restrict__ w2frag) {
  int t    = threadIdx.x;
  int lane = t & 31;
  int kt   = t >> 5;
  int nn   = lane & 15, hi = lane >> 4;
  uint32_t* dst = (uint32_t*)(w2frag + kt * 1024) + lane * 8;
  for (int v = 0; v < 8; ++v) {
    int half = v >> 2, vk = v & 3;
    uint32_t packed = 0;
    for (int b = 0; b < 4; ++b) {
      int k = kt * 64 + half * 32 + (vk >> 1) * 16 + hi * 8 + (vk & 1) * 4 + b;
      uint32_t sb = 0;
      if (nn < CDIM && k < HDIM)
        sb = (W2[(size_t)nn * HDIM + k] >= 0.f) ? 0x01u : 0xFFu;
      packed |= sb << (8 * b);
    }
    dst[v] = packed;
  }
}

__global__ void prep_bn_kernel(const float* __restrict__ b1,
                               const float* __restrict__ gamma,
                               const float* __restrict__ beta,
                               const float* __restrict__ rmean,
                               const float* __restrict__ rvar,
                               const float* __restrict__ b2,
                               float* __restrict__ sArr,
                               float* __restrict__ tArr,
                               int* __restrict__ b2bin) {
  int n = threadIdx.x;
  if (n < HPAD) {
    float s = 0.f, t2 = 0.f;
    if (n < HDIM) {
      s = gamma[n] * rsqrtf(rvar[n] + 1e-5f);
      float b1b = (b1[n] >= 0.f) ? 1.f : -1.f;
      t2 = beta[n] - rmean[n] * s + s * b1b;
    }
    sArr[n] = s;
    tArr[n] = t2;
  }
  if (n < 16) b2bin[n] = (n < CDIM) ? ((b2[n] >= 0.f) ? 1 : -1) : 0;
}

__global__ void __launch_bounds__(256)
bnn_fused_kernel(const float* __restrict__ x,
                 const uint8_t* __restrict__ w1frag,
                 const uint8_t* __restrict__ w2frag,
                 const float* __restrict__ sArr,
                 const float* __restrict__ tArr,
                 const int* __restrict__ b2bin,
                 float* __restrict__ out) {
  __shared__ uint8_t Abuf[2 * 64 * 1024];
  __shared__ uint8_t Hbuf[2 * 8 * 1024];

  const int tid  = threadIdx.x;
  const int lane = tid & 31;
  const int wave = tid >> 5;
  const size_t rowBase = (size_t)blockIdx.x * M_TILE;

  {
    const float4* xr = (const float4*)(x + rowBase * D_IN);
    uint32_t* ab = (uint32_t*)Abuf;
    for (int it = 0; it < 128; ++it) {
      int idx = it * 256 + tid;
      int m   = idx >> 10;
      int k4  = idx & 1023;
      float4 v = xr[(size_t)m * 1024 + k4];
      uint32_t packed =
            ((v.x >= 0.f) ? 0x01u : 0xFFu)
          | (((v.y >= 0.f) ? 0x01u : 0xFFu) << 8)
          | (((v.z >= 0.f) ? 0x01u : 0xFFu) << 16)
          | (((v.w >= 0.f) ? 0x01u : 0xFFu) << 24);
      int k    = k4 << 2;
      int k64  = k & 63;
      int half = k64 >> 5;
      int k32  = k64 & 31;
      int alane = (((k32 >> 3) & 1) << 4) + (m & 15);
      int av    = half * 4 + ((k32 >> 4) & 1) * 2 + ((k32 >> 2) & 1);
      int tile  = ((m >> 4) << 6) + (k >> 6);
      ab[tile * 256 + alane * 8 + av] = packed;
    }
  }
  __syncthreads();

  const int msub = wave >> 2;
  const int ngrp = wave & 3;
  for (int nti = 0; nti < 8; ++nti) {
    const int nt = ngrp * 8 + nti;
    v8i acc = {};
    const int4* abase = (const int4*)(Abuf + msub * 64 * 1024) + lane * 2;
    const int4* bbase = (const int4*)(w1frag + (size_t)nt * 64 * 1024) + lane * 2;
    for (int kt = 0; kt < 64; ++kt) {
      int4 a0 = abase[kt * 64 + 0];
      int4 a1 = abase[kt * 64 + 1];
      int4 b0 = bbase[kt * 64 + 0];
      int4 b1 = bbase[kt * 64 + 1];
      v8i av, bv;
      av[0] = a0.x; av[1] = a0.y; av[2] = a0.z; av[3] = a0.w;
      av[4] = a1.x; av[5] = a1.y; av[6] = a1.z; av[7] = a1.w;
      bv[0] = b0.x; bv[1] = b0.y; bv[2] = b0.z; bv[3] = b0.w;
      bv[4] = b1.x; bv[5] = b1.y; bv[6] = b1.z; bv[7] = b1.w;
      acc = __builtin_amdgcn_wmma_i32_16x16x64_iu8(
                true, av, true, bv, acc, false, false);
    }
    int nn = lane & 15;
    int n  = nt * 16 + nn;
    float sv = sArr[n];
    float tv = tArr[n];
    int mhiBase = (lane >> 4) * 8;
    int k64  = n & 63;
    int half = k64 >> 5;
    int k32  = k64 & 31;
    int hv   = half * 4 + ((k32 >> 4) & 1) * 2 + ((k32 >> 2) & 1);
    int hb   = k32 & 3;
    int hhi  = ((k32 >> 3) & 1) << 4;
    int htile = msub * 8 + (n >> 6);
    for (int r = 0; r < 8; ++r) {
      float z = fmaf(sv, (float)acc[r], tv);
      uint8_t sb = (z >= 0.f) ? 0x01u : 0xFFu;
      int hlane = hhi + mhiBase + r;
      Hbuf[htile * 1024 + hlane * 32 + hv * 4 + hb] = sb;
    }
  }
  __syncthreads();

  if (ngrp == 0) {
    v8i acc2 = {};
    const int4* habase = (const int4*)(Hbuf + msub * 8 * 1024) + lane * 2;
    const int4* wbase  = (const int4*)w2frag + lane * 2;
    for (int kt = 0; kt < 8; ++kt) {
      int4 a0 = habase[kt * 64 + 0];
      int4 a1 = habase[kt * 64 + 1];
      int4 b0 = wbase[kt * 64 + 0];
      int4 b1 = wbase[kt * 64 + 1];
      v8i av, bv;
      av[0] = a0.x; av[1] = a0.y; av[2] = a0.z; av[3] = a0.w;
      av[4] = a1.x; av[5] = a1.y; av[6] = a1.z; av[7] = a1.w;
      bv[0] = b0.x; bv[1] = b0.y; bv[2] = b0.z; bv[3] = b0.w;
      bv[4] = b1.x; bv[5] = b1.y; bv[6] = b1.z; bv[7] = b1.w;
      acc2 = __builtin_amdgcn_wmma_i32_16x16x64_iu8(
                 true, av, true, bv, acc2, false, false);
    }
    int c = lane & 15;
    if (c < CDIM) {
      int bb = b2bin[c];
      int mhiBase = (lane >> 4) * 8;
      for (int r = 0; r < 8; ++r) {
        size_t m = rowBase + msub * 16 + mhiBase + r;
        out[m * CDIM + c] = (float)(acc2[r] + bb);
      }
    }
  }
}

extern "C" void kernel_launch(void* const* d_in, const int* in_sizes, int n_in,
                              void* d_out, int out_size, void* d_ws, size_t ws_size,
                              hipStream_t stream) {
  const float* x     = (const float*)d_in[0];
  const float* W1    = (const float*)d_in[1];
  const float* b1    = (const float*)d_in[2];
  const float* gamma = (const float*)d_in[3];
  const float* beta  = (const float*)d_in[4];
  const float* rmean = (const float*)d_in[5];
  const float* rvar  = (const float*)d_in[6];
  const float* W2    = (const float*)d_in[7];
  const float* b2    = (const float*)d_in[8];

  uint8_t* ws      = (uint8_t*)d_ws;
  uint8_t* w1frag  = ws + OFF_W1FRAG;
  uint8_t* w2frag  = ws + OFF_W2FRAG;
  float*   sArr    = (float*)(ws + OFF_S);
  float*   tArr    = (float*)(ws + OFF_T);
  int*     b2bin   = (int*)(ws + OFF_B2);
  float*   out     = (float*)d_out;

  pack_w1_kernel<<<256, 256, 0, stream>>>(W1, w1frag);
  pack_w2_kernel<<<1, 256, 0, stream>>>(W2, w2frag);
  prep_bn_kernel<<<1, 512, 0, stream>>>(b1, gamma, beta, rmean, rvar, b2,
                                        sArr, tArr, b2bin);
  bnn_fused_kernel<<<B_ROWS / M_TILE, 256, 0, stream>>>(
      x, w1frag, w2frag, sArr, tArr, b2bin, out);
}
